// GeometricallyAlignedAttentionLayer_84619445666678
// MI455X (gfx1250) — matrix operand order repaired and checked by reading, not compiled, not run on hardware
//
#include <hip/hip_runtime.h>
#include <hip/hip_bf16.h>

// ---------------------------------------------------------------------------
// Problem constants (from reference)
// ---------------------------------------------------------------------------
#define NB    2
#define SEQ   2048
#define DM    512
#define KNNK  16
#define FDIM  2048
#define LN_EPS 1e-6f

typedef float v2f  __attribute__((ext_vector_type(2)));
typedef float v8f  __attribute__((ext_vector_type(8)));
typedef __bf16 v16bf __attribute__((ext_vector_type(16)));
typedef int   v4i_ __attribute__((ext_vector_type(4)));

// ---------------------------------------------------------------------------
// CDNA5 async global->LDS copy support (probe-guarded)
// ---------------------------------------------------------------------------
#if defined(__gfx1250__) && \
    __has_builtin(__builtin_amdgcn_global_load_async_to_lds_b128) && \
    __has_builtin(__builtin_amdgcn_s_wait_asynccnt)
#define USE_ASYNC_LDS 1
#else
#define USE_ASYNC_LDS 0
#endif

#if USE_ASYNC_LDS
typedef __attribute__((address_space(1))) v4i_ g_v4i;   // global (AS1)
typedef __attribute__((address_space(3))) v4i_ l_v4i;   // LDS (AS3)
__device__ __forceinline__ void async_b128(const void* g, const void* l) {
    // flat LDS address: low 32 bits are the LDS byte offset (ISA 10.2)
    __builtin_amdgcn_global_load_async_to_lds_b128(
        (g_v4i*)(unsigned long long)g,
        (l_v4i*)(unsigned long long)(unsigned)(unsigned long long)l,
        0, 0);
}
__device__ __forceinline__ void async_wait0() {
    __builtin_amdgcn_s_wait_asynccnt(0);
}
#endif

// ---------------------------------------------------------------------------
// helpers
// ---------------------------------------------------------------------------
__device__ __forceinline__ unsigned short f2bf(float f) {
    unsigned int u = __builtin_bit_cast(unsigned int, f);
    unsigned int r = u + 0x7FFFu + ((u >> 16) & 1u);
    return (unsigned short)(r >> 16);
}

__device__ __forceinline__ float wave_sum(float v) {
#pragma unroll
    for (int m = 16; m >= 1; m >>= 1) v += __shfl_xor(v, m, 32);
    return v;
}

__device__ __forceinline__ float block_sum(float v, float* red) {
    int t = threadIdx.x;
    red[t] = v;
    __syncthreads();
#pragma unroll
    for (int s = 128; s > 0; s >>= 1) {
        if (t < s) red[t] += red[t + s];
        __syncthreads();
    }
    float r = red[0];
    __syncthreads();
    return r;
}

__device__ __forceinline__ float gelu_tanh(float x) {
    float x3 = x * x * x;
    return 0.5f * x * (1.0f + tanhf(0.7978845608028654f * (x + 0.044715f * x3)));
}

__device__ __forceinline__ void topk_insert(float (&a)[KNNK], int (&ix)[KNNK],
                                            float v, int j) {
    if (v >= a[KNNK - 1]) return;
    a[KNNK - 1] = v;
    ix[KNNK - 1] = j;
#pragma unroll
    for (int t = KNNK - 1; t > 0; --t) {
        if (a[t] < a[t - 1]) {
            float tf = a[t]; a[t] = a[t - 1]; a[t - 1] = tf;
            int   ti = ix[t]; ix[t] = ix[t - 1]; ix[t - 1] = ti;
        }
    }
}

// ---------------------------------------------------------------------------
// weight convert + transpose: Wt[n*K + k] = bf16(W[k*N + n])
// ---------------------------------------------------------------------------
__global__ void wtrans_kernel(const float* __restrict__ W,
                              unsigned short* __restrict__ Wt, int K, int N) {
    int idx = blockIdx.x * 256 + threadIdx.x;   // index into Wt (N*K)
    if (idx < K * N) {
        int n = idx / K;
        int k = idx - n * K;
        Wt[idx] = f2bf(W[(size_t)k * N + n]);
    }
}

// ---------------------------------------------------------------------------
// LayerNorm over D=512, 256 threads per row, bf16 output
// ---------------------------------------------------------------------------
__global__ __launch_bounds__(256) void ln_kernel(const float* __restrict__ X,
                                                 const float* __restrict__ g,
                                                 const float* __restrict__ be,
                                                 unsigned short* __restrict__ out) {
    __shared__ float red[256];
    int row = blockIdx.x;
    int t = threadIdx.x;
    const float* x = X + (size_t)row * DM;
    float a = x[t], b = x[t + 256];
    float s = block_sum(a + b, red);
    float mean = s * (1.0f / DM);
    float da = a - mean, db = b - mean;
    float v = block_sum(da * da + db * db, red);
    float inv = rsqrtf(v * (1.0f / DM) + LN_EPS);
    out[(size_t)row * DM + t]       = f2bf(g[t] * da * inv + be[t]);
    out[(size_t)row * DM + t + 256] = f2bf(g[t + 256] * db * inv + be[t + 256]);
}

// ---------------------------------------------------------------------------
// p2 = sum(positions^2) per row
// ---------------------------------------------------------------------------
__global__ __launch_bounds__(256) void p2_kernel(const float* __restrict__ pos,
                                                 float* __restrict__ p2g) {
    __shared__ float red[256];
    int row = blockIdx.x;
    int t = threadIdx.x;
    const float* x = pos + (size_t)row * DM;
    float a = x[t], b = x[t + 256];
    float s = block_sum(a * a + b * b, red);
    if (t == 0) p2g[row] = s;
}

// ---------------------------------------------------------------------------
// bf16 WMMA GEMM: out = act(A[MxK] @ W[KxN] + bias) (+ residual)
// W is pre-transposed: Wt[N x K] row-major, so both tiles stage contiguously.
// 64x128 tile / workgroup, 8 waves; each wave owns a 32x32 output patch
// (2 A frags x 2 B frags -> 4 WMMAs per 32-wide K step).
// Double-buffered LDS; async global->LDS copies on the CDNA5 ASYNCcnt path.
// ---------------------------------------------------------------------------
__global__ __launch_bounds__(256) void gemm_bf16_kernel(
    const unsigned short* __restrict__ A, const unsigned short* __restrict__ Wt,
    const float* __restrict__ bias, const float* __restrict__ resid,
    float* __restrict__ outF, unsigned short* __restrict__ outB,
    int M, int Nn, int K, int gelu_flag) {
    __shared__ __align__(16) unsigned short As[2][64 * 32];    // M x K tiles
    __shared__ __align__(16) unsigned short Bs[2][128 * 32];   // N x K tiles

    const int tid  = threadIdx.x;
    const int w    = tid >> 5;
    const int lane = tid & 31;
    const int ml   = lane & 15;
    const int half = lane >> 4;
    const int bm   = blockIdx.y * 64;
    const int bn   = blockIdx.x * 128;
    const int rM   = (w >> 2) * 32;    // wave row base (0 / 32)
    const int cN   = (w & 3) * 32;     // wave col base (0/32/64/96)

    // staging slots: A = 1x b128/thread, B = 2x b128/thread
    const int arow = tid >> 2, ak = (tid & 3) * 8;
    v8f acc[2][2];
#pragma unroll
    for (int s = 0; s < 2; ++s)
#pragma unroll
        for (int t = 0; t < 2; ++t)
            acc[s][t] = (v8f){0.f, 0.f, 0.f, 0.f, 0.f, 0.f, 0.f, 0.f};

#if USE_ASYNC_LDS
    // ---- prologue: async-stage tile kk=0 into buffer 0 ----
    async_b128(A + (size_t)(bm + arow) * K + ak, &As[0][arow * 32 + ak]);
#pragma unroll
    for (int s = 0; s < 2; ++s) {
        int flat = tid + 256 * s;
        int br = flat >> 2, bkk = (flat & 3) * 8;
        async_b128(Wt + (size_t)(bn + br) * K + bkk, &Bs[0][br * 32 + bkk]);
    }
    async_wait0();
    __syncthreads();
    int buf = 0;
    for (int kk = 0; kk < K; kk += 32) {
        if (kk + 32 < K) {
            async_b128(A + (size_t)(bm + arow) * K + kk + 32 + ak,
                       &As[buf ^ 1][arow * 32 + ak]);
#pragma unroll
            for (int s = 0; s < 2; ++s) {
                int flat = tid + 256 * s;
                int br = flat >> 2, bkk = (flat & 3) * 8;
                async_b128(Wt + (size_t)(bn + br) * K + kk + 32 + bkk,
                           &Bs[buf ^ 1][br * 32 + bkk]);
            }
        }
        union Frag { unsigned int u[8]; v16bf v; } fa[2], fb[2];
#pragma unroll
        for (int j = 0; j < 8; ++j) {
            int kj = (j < 4 ? 2 * j : 16 + 2 * (j - 4)) + 8 * half;
#pragma unroll
            for (int s = 0; s < 2; ++s) {
                fa[s].u[j] = *(const unsigned int*)&As[buf][(rM + s * 16 + ml) * 32 + kj];
                fb[s].u[j] = *(const unsigned int*)&Bs[buf][(cN + s * 16 + ml) * 32 + kj];
            }
        }
#pragma unroll
        for (int s = 0; s < 2; ++s)
#pragma unroll
            for (int t = 0; t < 2; ++t)
                acc[s][t] = __builtin_amdgcn_wmma_f32_16x16x32_bf16(
                    false, fa[s].v, false, fb[t].v, (short)0, acc[s][t], false, false);
        async_wait0();
        __syncthreads();
        buf ^= 1;
    }
#else
    for (int kk = 0; kk < K; kk += 32) {
        uint4 ra = *(const uint4*)(A + (size_t)(bm + arow) * K + kk + ak);
        uint4 rb[2];
#pragma unroll
        for (int s = 0; s < 2; ++s) {
            int flat = tid + 256 * s;
            int br = flat >> 2, bkk = (flat & 3) * 8;
            rb[s] = *(const uint4*)(Wt + (size_t)(bn + br) * K + kk + bkk);
        }
        __syncthreads();
        *(uint4*)&As[0][arow * 32 + ak] = ra;
#pragma unroll
        for (int s = 0; s < 2; ++s) {
            int flat = tid + 256 * s;
            int br = flat >> 2, bkk = (flat & 3) * 8;
            *(uint4*)&Bs[0][br * 32 + bkk] = rb[s];
        }
        __syncthreads();
        union Frag { unsigned int u[8]; v16bf v; } fa[2], fb[2];
#pragma unroll
        for (int j = 0; j < 8; ++j) {
            int kj = (j < 4 ? 2 * j : 16 + 2 * (j - 4)) + 8 * half;
#pragma unroll
            for (int s = 0; s < 2; ++s) {
                fa[s].u[j] = *(const unsigned int*)&As[0][(rM + s * 16 + ml) * 32 + kj];
                fb[s].u[j] = *(const unsigned int*)&Bs[0][(cN + s * 16 + ml) * 32 + kj];
            }
        }
#pragma unroll
        for (int s = 0; s < 2; ++s)
#pragma unroll
            for (int t = 0; t < 2; ++t)
                acc[s][t] = __builtin_amdgcn_wmma_f32_16x16x32_bf16(
                    false, fa[s].v, false, fb[t].v, (short)0, acc[s][t], false, false);
    }
#endif

    // epilogue: C layout -> lane n = lane&15, VGPR r -> m = r + 8*(lane>>4)
#pragma unroll
    for (int t = 0; t < 2; ++t) {
        const int gn = bn + cN + t * 16 + ml;
        const float bv = bias ? bias[gn] : 0.0f;
#pragma unroll
        for (int s = 0; s < 2; ++s) {
#pragma unroll
            for (int r = 0; r < 8; ++r) {
                int gm = bm + rM + s * 16 + r + 8 * half;
                float v = acc[s][t][r] + bv;
                size_t o = (size_t)gm * Nn + gn;
                if (resid) v += resid[o];
                if (gelu_flag) v = gelu_tanh(v);
                if (outF) outF[o] = v;
                if (outB) outB[o] = f2bf(v);
            }
        }
    }
}

// ---------------------------------------------------------------------------
// Hyperbolic distance + top-16 kernel (f32 WMMA Gram matrix).
// WG = 128 threads (4 waves), 16 queries per WG; each wave streams 16-key
// tiles. Ranking uses the monotone surrogate arg = 1 + 2c*diff2/den.
// ---------------------------------------------------------------------------
__global__ __launch_bounds__(128) void topk_kernel(
    const float* __restrict__ pos, const float* __restrict__ p2g,
    const float* __restrict__ cptr, int* __restrict__ idxout) {
    __shared__ __align__(16) float qpos[16 * DM];        // 32 KB
    __shared__ __align__(16) float kch[4][16 * 64];      // 16 KB (per-wave chunk)
    __shared__ float tilec[4][16 * 16];                  // 4 KB
    __shared__ float marg[4][16][KNNK];                  // 4 KB
    __shared__ int   midx[4][16][KNNK];                  // 4 KB

    const int b = blockIdx.y;
    const int qBase = blockIdx.x * 16;
    const int tid = threadIdx.x;
    const int w = tid >> 5;
    const int lane = tid & 31;
    const int ml = lane & 15;
    const int half = lane >> 4;
    const float c = cptr[0];
    const float* posb = pos + (size_t)b * SEQ * DM;

#pragma unroll
    for (int i = 0; i < 16; ++i) {
        int flat = tid + 128 * i;          // 0..2047 float4 slots
        int q = flat >> 7;                 // 128 float4 per row
        int c4 = flat & 127;
        ((float4*)qpos)[q * 128 + c4] =
            ((const float4*)(posb + (size_t)(qBase + q) * DM))[c4];
    }
    __syncthreads();

    float barg[KNNK];
    int bidx[KNNK];
#pragma unroll
    for (int t = 0; t < KNNK; ++t) { barg[t] = 3.0e38f; bidx[t] = 0; }
    const float p2i = p2g[b * SEQ + qBase + ml];

    for (int tile = 0; tile < SEQ / 64; ++tile) {
        const int keyBase = tile * 64 + w * 16;
        v8f acc = {0.f, 0.f, 0.f, 0.f, 0.f, 0.f, 0.f, 0.f};

        for (int kc = 0; kc < DM; kc += 64) {
#pragma unroll
            for (int i = 0; i < 8; ++i) {
                int flat = lane + 32 * i;  // 0..255 float4 slots
                int key = flat >> 4;
                int c4 = flat & 15;
                ((float4*)(kch[w]))[key * 16 + c4] =
                    ((const float4*)(posb + (size_t)(keyBase + key) * DM + kc))[c4];
            }
#pragma unroll
            for (int ks = 0; ks < 16; ++ks) {
                v2f a = *(const v2f*)&qpos[ml * DM + kc + ks * 4 + 2 * half];
                v2f bb = *(const v2f*)&kch[w][ml * 64 + ks * 4 + 2 * half];
                acc = __builtin_amdgcn_wmma_f32_16x16x4_f32(false, a, false, bb,
                                                            (short)0, acc, false, false);
            }
        }
#pragma unroll
        for (int r = 0; r < 8; ++r)
            tilec[w][(r + 8 * half) * 16 + ml] = acc[r];

        if (lane < 16) {
            const int m = lane;
#pragma unroll 1
            for (int n = 0; n < 16; ++n) {
                int j = keyBase + n;
                float ip = tilec[w][m * 16 + n];
                float p2j = p2g[b * SEQ + j];
                float diff = fmaxf(p2i + p2j - 2.0f * ip, 0.0f);
                float den = fmaxf((1.0f - c * p2i) * (1.0f - c * p2j), 1e-8f);
                float arg = 1.0f + 2.0f * c * diff / den;
                topk_insert(barg, bidx, arg, j);
            }
        }
    }

    if (lane < 16) {
#pragma unroll
        for (int t = 0; t < KNNK; ++t) {
            marg[w][lane][t] = barg[t];
            midx[w][lane][t] = bidx[t];
        }
    }
    __syncthreads();

    if (w == 0 && lane < 16) {
        float fa[KNNK]; int fi[KNNK];
#pragma unroll
        for (int t = 0; t < KNNK; ++t) { fa[t] = 3.0e38f; fi[t] = 0; }
        for (int ww = 0; ww < 4; ++ww)
#pragma unroll 1
            for (int t = 0; t < KNNK; ++t)
                topk_insert(fa, fi, marg[ww][lane][t], midx[ww][lane][t]);
        int* dst = idxout + ((size_t)(b * SEQ + qBase + lane)) * KNNK;
#pragma unroll
        for (int t = 0; t < KNNK; ++t) dst[t] = fi[t];
    }
}

// ---------------------------------------------------------------------------
// Per-(b,n,k) wave kernel: mobius_add(-qp, kp, c) -> logmap0 -> LN(geo) ->
// align = q . LN(geo), feat = q . k[j]; scores out.
// ---------------------------------------------------------------------------
__global__ __launch_bounds__(256) void geo_score_kernel(
    const float* __restrict__ pos, const float* __restrict__ p2g,
    const float* __restrict__ qbuf, const float* __restrict__ kbuf,
    const int* __restrict__ idxb, const float* __restrict__ cptr,
    const float* __restrict__ geo_s, const float* __restrict__ geo_b,
    const float* __restrict__ align_scale, const float* __restrict__ feat_scale,
    float* __restrict__ scores) {
    const int task = blockIdx.x * 8 + (threadIdx.x >> 5);
    const int lane = threadIdx.x & 31;
    const int b = task / (SEQ * KNNK);
    const int rem = task - b * (SEQ * KNNK);
    const int n = rem / KNNK;
    const int j = idxb[task];

    const float c = cptr[0];
    const float sqrtc = sqrtf(c);
    const float* qp = pos + ((size_t)b * SEQ + n) * DM;
    const float* kp = pos + ((size_t)b * SEQ + j) * DM;
    const float* qv = qbuf + ((size_t)b * SEQ + n) * DM;
    const float* kv = kbuf + ((size_t)b * SEQ + j) * DM;
    const float p2i = p2g[b * SEQ + n];
    const float p2j = p2g[b * SEQ + j];

    float qpr[16], kpr[16], mreg[16];
    float dot = 0.0f;
#pragma unroll
    for (int i = 0; i < 16; ++i) {
        int d = lane + 32 * i;
        qpr[i] = qp[d];
        kpr[i] = kp[d];
        dot += qpr[i] * kpr[i];
    }
    dot = wave_sum(dot);

    const float xy = -dot;                       // x = -query_pos, y = key_pos
    const float coefx = 1.0f + 2.0f * c * xy + c * p2j;
    const float coefy = 1.0f - c * p2i;
    float den = 1.0f + 2.0f * c * xy + c * c * p2i * p2j;
    den = fmaxf(den, 1e-8f);
    const float inv_den = 1.0f / den;

    float nrm2 = 0.0f;
#pragma unroll
    for (int i = 0; i < 16; ++i) {
        float m = (coefx * (-qpr[i]) + coefy * kpr[i]) * inv_den;
        mreg[i] = m;
        nrm2 += m * m;
    }
    nrm2 = wave_sum(nrm2);
    const float ynorm = fmaxf(sqrtf(nrm2), 1e-8f);
    const float at = fminf(sqrtc * ynorm, 1.0f - 1e-7f);
    const float scale = atanhf(at) / (sqrtc * ynorm);

    float sum = 0.0f, sumsq = 0.0f;
#pragma unroll
    for (int i = 0; i < 16; ++i) {
        float g = scale * mreg[i];
        mreg[i] = g;
        sum += g;
        sumsq += g * g;
    }
    sum = wave_sum(sum);
    sumsq = wave_sum(sumsq);
    const float mean = sum * (1.0f / DM);
    const float var = sumsq * (1.0f / DM) - mean * mean;
    const float inv = rsqrtf(var + LN_EPS);

    float align = 0.0f, feat = 0.0f;
#pragma unroll
    for (int i = 0; i < 16; ++i) {
        int d = lane + 32 * i;
        float gl = geo_s[d] * (mreg[i] - mean) * inv + geo_b[d];
        float qd = qv[d];
        align += qd * gl;
        feat += qd * kv[d];
    }
    align = wave_sum(align);
    feat = wave_sum(feat);
    if (lane == 0)
        scores[task] = feat_scale[0] * feat + align_scale[0] * align;
}

// ---------------------------------------------------------------------------
// Softmax over 16 scores + weighted gather of V -> bf16 attn rows
// ---------------------------------------------------------------------------
__global__ __launch_bounds__(256) void attn_kernel(
    const float* __restrict__ scores, const int* __restrict__ idxb,
    const float* __restrict__ vbuf, unsigned short* __restrict__ attn_bf) {
    __shared__ float sc[KNNK];
    __shared__ int si[KNNK];
    const int bn = blockIdx.x;
    const int t = threadIdx.x;
    if (t < KNNK) {
        sc[t] = scores[bn * KNNK + t];
        si[t] = idxb[bn * KNNK + t];
    }
    __syncthreads();

    float mx = -3.0e38f;
#pragma unroll
    for (int k = 0; k < KNNK; ++k) mx = fmaxf(mx, sc[k]);
    float wk[KNNK];
    float se = 0.0f;
#pragma unroll
    for (int k = 0; k < KNNK; ++k) {
        wk[k] = __expf(sc[k] - mx);
        se += wk[k];
    }
    const float inv = 1.0f / se;
    const int b = bn / SEQ;
    const size_t base = (size_t)b * SEQ * DM;
#pragma unroll
    for (int d = t; d < DM; d += 256) {
        float a = 0.0f;
#pragma unroll
        for (int k = 0; k < KNNK; ++k)
            a += wk[k] * vbuf[base + (size_t)si[k] * DM + d];
        attn_bf[(size_t)bn * DM + d] = f2bf(a * inv);
    }
}

// ---------------------------------------------------------------------------
// launch
// ---------------------------------------------------------------------------
extern "C" void kernel_launch(void* const* d_in, const int* in_sizes, int n_in,
                              void* d_out, int out_size, void* d_ws, size_t ws_size,
                              hipStream_t stream) {
    const float* x        = (const float*)d_in[0];
    const float* positions= (const float*)d_in[1];
    const float* c        = (const float*)d_in[2];
    const float* Wq = (const float*)d_in[3];  const float* bq = (const float*)d_in[4];
    const float* Wk = (const float*)d_in[5];  const float* bk = (const float*)d_in[6];
    const float* Wv = (const float*)d_in[7];  const float* bv = (const float*)d_in[8];
    const float* Wo = (const float*)d_in[9];  const float* bo = (const float*)d_in[10];
    const float* W1 = (const float*)d_in[11]; const float* b1 = (const float*)d_in[12];
    const float* W2 = (const float*)d_in[13]; const float* b2 = (const float*)d_in[14];
    const float* ln1_s = (const float*)d_in[15]; const float* ln1_b = (const float*)d_in[16];
    const float* ln2_s = (const float*)d_in[17]; const float* ln2_b = (const float*)d_in[18];
    const float* geo_s = (const float*)d_in[19]; const float* geo_b = (const float*)d_in[20];
    const float* align_scale = (const float*)d_in[21];
    const float* feat_scale  = (const float*)d_in[22];

    const int ROWS = NB * SEQ;               // 4096

    // ---- workspace carve (bytes, 256-aligned); ~66 MB total ----
    char* wsp = (char*)d_ws;
    size_t off = 0;
    auto alloc = [&](size_t bytes) {
        void* p = wsp + off;
        off = (off + bytes + 255) & ~(size_t)255;
        return p;
    };
    unsigned short* xn_bf   = (unsigned short*)alloc((size_t)ROWS * DM * 2);
    unsigned short* h_bf    = (unsigned short*)alloc((size_t)ROWS * DM * 2);
    unsigned short* attn_bf = (unsigned short*)alloc((size_t)ROWS * DM * 2);
    unsigned short* t_bf    = (unsigned short*)alloc((size_t)ROWS * FDIM * 2);
    float* q_f  = (float*)alloc((size_t)ROWS * DM * 4);
    float* k_f  = (float*)alloc((size_t)ROWS * DM * 4);
    float* v_f  = (float*)alloc((size_t)ROWS * DM * 4);
    float* x1_f = (float*)alloc((size_t)ROWS * DM * 4);
    float* p2g  = (float*)alloc((size_t)ROWS * 4);
    int*   idxb = (int*)alloc((size_t)ROWS * KNNK * 4);
    float* scores = (float*)alloc((size_t)ROWS * KNNK * 4);
    unsigned short* Wq_t = (unsigned short*)alloc((size_t)DM * DM * 2);
    unsigned short* Wk_t = (unsigned short*)alloc((size_t)DM * DM * 2);
    unsigned short* Wv_t = (unsigned short*)alloc((size_t)DM * DM * 2);
    unsigned short* Wo_t = (unsigned short*)alloc((size_t)DM * DM * 2);
    unsigned short* W1_t = (unsigned short*)alloc((size_t)DM * FDIM * 2);
    unsigned short* W2_t = (unsigned short*)alloc((size_t)FDIM * DM * 2);
    (void)ws_size; (void)n_in; (void)in_sizes; (void)out_size;

    // ---- 1) weight convert + transpose to bf16 Wt[N x K] ----
    auto wt = [&](const float* src, unsigned short* dst, int K, int N) {
        int n = K * N;
        wtrans_kernel<<<(n + 255) / 256, 256, 0, stream>>>(src, dst, K, N);
    };
    wt(Wq, Wq_t, DM, DM);
    wt(Wk, Wk_t, DM, DM);
    wt(Wv, Wv_t, DM, DM);
    wt(Wo, Wo_t, DM, DM);
    wt(W1, W1_t, DM, FDIM);
    wt(W2, W2_t, FDIM, DM);

    // ---- 2) LN1 and p2 ----
    ln_kernel<<<ROWS, 256, 0, stream>>>(x, ln1_s, ln1_b, xn_bf);
    p2_kernel<<<ROWS, 256, 0, stream>>>(positions, p2g);

    // ---- 3) Q/K/V projections (bf16 WMMA, f32 out) ----
    {
        dim3 grid(DM / 128, ROWS / 64);
        gemm_bf16_kernel<<<grid, 256, 0, stream>>>(xn_bf, Wq_t, bq, nullptr,
                                                   q_f, nullptr, ROWS, DM, DM, 0);
        gemm_bf16_kernel<<<grid, 256, 0, stream>>>(xn_bf, Wk_t, bk, nullptr,
                                                   k_f, nullptr, ROWS, DM, DM, 0);
        gemm_bf16_kernel<<<grid, 256, 0, stream>>>(xn_bf, Wv_t, bv, nullptr,
                                                   v_f, nullptr, ROWS, DM, DM, 0);
    }

    // ---- 4) hyperbolic distances + top-16 (f32 WMMA Gram) ----
    {
        dim3 grid(SEQ / 16, NB);
        topk_kernel<<<grid, 128, 0, stream>>>(positions, p2g, c, idxb);
    }

    // ---- 5) geo features + scores ----
    geo_score_kernel<<<(ROWS * KNNK) / 8, 256, 0, stream>>>(
        positions, p2g, q_f, k_f, idxb, c, geo_s, geo_b, align_scale, feat_scale,
        scores);

    // ---- 6) softmax + weighted V ----
    attn_kernel<<<ROWS, 256, 0, stream>>>(scores, idxb, v_f, attn_bf);

    // ---- 7) x1 = x + attn @ Wo + bo ----
    {
        dim3 grid(DM / 128, ROWS / 64);
        gemm_bf16_kernel<<<grid, 256, 0, stream>>>(attn_bf, Wo_t, bo, x,
                                                   x1_f, nullptr, ROWS, DM, DM, 0);
    }

    // ---- 8) LN2 ----
    ln_kernel<<<ROWS, 256, 0, stream>>>(x1_f, ln2_s, ln2_b, h_bf);

    // ---- 9) t = gelu(h @ W1 + b1)  (bf16 out only) ----
    {
        dim3 grid(FDIM / 128, ROWS / 64);
        gemm_bf16_kernel<<<grid, 256, 0, stream>>>(h_bf, W1_t, b1, nullptr,
                                                   nullptr, t_bf, ROWS, FDIM, DM, 1);
    }

    // ---- 10) out = x1 + t @ W2 + b2 ----
    {
        dim3 grid(DM / 128, ROWS / 64);
        gemm_bf16_kernel<<<grid, 256, 0, stream>>>(t_bf, W2_t, b2, x1_f,
                                                   (float*)d_out, nullptr,
                                                   ROWS, DM, FDIM, 0);
    }
}
